// AttentionTSSA_75453985457225
// MI455X (gfx1250) — compile-verified, matrix operand order checked by reading, not run on hardware
//
#include <hip/hip_runtime.h>
#include <hip/hip_bf16.h>
#include <cstdint>

typedef __attribute__((ext_vector_type(16))) __bf16 v16bf;
typedef __attribute__((ext_vector_type(8)))  float  v8f;

#define HEADS 16
#define BATCH 4
#define NTOK  4096
#define CDIM  1024
#define DDIM  64

// GEMM tiling: block 128x128x32, 8 waves (2x4), wave tile 64x32 -> 8 WMMA/K-step
#define BM 128
#define BN 128
#define BK 32
#define LDT 40   // padded LDS stride in bf16 elements (80B rows -> 16B aligned)

__device__ __forceinline__ unsigned short f2bf(float f) {
  unsigned int u = __float_as_uint(f);
  u += 0x7FFFu + ((u >> 16) & 1u);          // round-to-nearest-even
  return (unsigned short)(u >> 16);
}

// low 32 bits of a generic LDS address == wave-relative LDS byte offset (ISA 10.2)
__device__ __forceinline__ unsigned lds_off(const void* p) {
  return (unsigned)(uintptr_t)p;
}

__device__ __forceinline__ void async_copy_b128(unsigned lds_byte_off,
                                                const void* gaddr) {
  asm volatile("global_load_async_to_lds_b128 %0, %1, off"
               :: "v"(lds_byte_off), "v"(gaddr) : "memory");
}

// async loads complete in-order: cnt<=4 => the 4 ops of the older tile landed
__device__ __forceinline__ void wait_async_le4() {
  asm volatile("s_wait_asynccnt 0x4" ::: "memory");
}
__device__ __forceinline__ void wait_async0() {
  asm volatile("s_wait_asynccnt 0x0" ::: "memory");
}

// Y[m,o] = sum_k A[m,k] * Bw[o,k]   (A, Bw already bf16 in memory)
// MODE 0: scatter fp32 to (b,h,n,d).  MODE 1: Y[m,o] = acc + bias[o] to d_out.
template<int MODE>
__global__ __launch_bounds__(256)
void gemm_bf16_wmma(const unsigned short* __restrict__ A,
                    const unsigned short* __restrict__ Bw,
                    float* __restrict__ out, const float* __restrict__ bias,
                    int M, int N, int K) {
  __shared__ unsigned short sA[2][BM * LDT];
  __shared__ unsigned short sB[2][BN * LDT];

  const int tid  = threadIdx.x;
  const int lane = tid & 31;
  const int wave = tid >> 5;
  const int wm   = wave >> 2;            // 0..1 : 64 rows each
  const int wn   = wave & 3;             // 0..3 : 32 cols each
  const int m0   = blockIdx.y * BM;
  const int n0   = blockIdx.x * BN;

  // async staging map: wave issues 2 x B128 per matrix per K-tile
  // lane covers row = wave*16 + j*8 + (lane>>2), 8 bf16 elems at col (lane&3)*8
  const int colel = (lane & 3) * 8;
  const int row0  = wave * 16 + (lane >> 2);

  const int lrow = lane & 15;
  const int klo  = (lane >> 4) * 8;      // ISA bf16 A/B frag layout: 0 or 8

  v8f acc[4][2] = {};

  const unsigned short* gA = A  + (size_t)(m0 + row0) * K + colel;
  const unsigned short* gB = Bw + (size_t)(n0 + row0) * K + colel;
  unsigned ldsA0[2], ldsA1[2], ldsB0[2], ldsB1[2];
  #pragma unroll
  for (int s = 0; s < 2; ++s) {
    ldsA0[s] = lds_off(&sA[s][(row0    ) * LDT + colel]);
    ldsA1[s] = lds_off(&sA[s][(row0 + 8) * LDT + colel]);
    ldsB0[s] = lds_off(&sB[s][(row0    ) * LDT + colel]);
    ldsB1[s] = lds_off(&sB[s][(row0 + 8) * LDT + colel]);
  }

  // prologue: stage tile 0 into buffer 0
  async_copy_b128(ldsA0[0], gA);
  async_copy_b128(ldsA1[0], gA + (size_t)8 * K);
  async_copy_b128(ldsB0[0], gB);
  async_copy_b128(ldsB1[0], gB + (size_t)8 * K);

  for (int k0 = 0, it = 0; k0 < K; k0 += BK, ++it) {
    const int cur = it & 1;
    const int kn  = k0 + BK;
    if (kn < K) {  // prefetch next tile into the other buffer (overlaps compute)
      const int nxt = cur ^ 1;
      async_copy_b128(ldsA0[nxt], gA + kn);
      async_copy_b128(ldsA1[nxt], gA + kn + (size_t)8 * K);
      async_copy_b128(ldsB0[nxt], gB + kn);
      async_copy_b128(ldsB1[nxt], gB + kn + (size_t)8 * K);
      wait_async_le4();                  // tile k0 landed; tile kn still in flight
    } else {
      wait_async0();
    }
    __syncthreads();                     // all waves' copies of tile k0 visible

    // ---- per-lane fragment loads (2x ds_load_b128 each) + 8 WMMAs ----
    v16bf afrag[4];
    #pragma unroll
    for (int fm = 0; fm < 4; ++fm) {
      int row = wm * 64 + fm * 16 + lrow;
      union { v16bf v; uint4 q[2]; } u;
      u.q[0] = *(const uint4*)&sA[cur][row * LDT + klo];       // K = klo..klo+7
      u.q[1] = *(const uint4*)&sA[cur][row * LDT + klo + 16];  // K = klo+16..+23
      afrag[fm] = u.v;
    }
    #pragma unroll
    for (int fn = 0; fn < 2; ++fn) {
      int col = wn * 32 + fn * 16 + lrow;
      union { v16bf v; uint4 q[2]; } u;
      u.q[0] = *(const uint4*)&sB[cur][col * LDT + klo];
      u.q[1] = *(const uint4*)&sB[cur][col * LDT + klo + 16];
      v16bf bfrag = u.v;
      #pragma unroll
      for (int fm = 0; fm < 4; ++fm) {
        acc[fm][fn] = __builtin_amdgcn_wmma_f32_16x16x32_bf16(
            false, afrag[fm], false, bfrag, (short)0, acc[fm][fn], false, false);
      }
    }
    __syncthreads();                     // reads done: buffer reusable at it+2
  }

  // ---- epilogue: C layout lane0-15 -> N=lane,M=r ; lane16-31 -> M=8+r ----
  #pragma unroll
  for (int fm = 0; fm < 4; ++fm) {
    #pragma unroll
    for (int fn = 0; fn < 2; ++fn) {
      int mbase = m0 + wm * 64 + fm * 16 + ((lane >> 4) * 8);
      int o     = n0 + wn * 32 + fn * 16 + lrow;
      #pragma unroll
      for (int r = 0; r < 8; ++r) {
        int m = mbase + r;
        float v = acc[fm][fn][r];
        if (MODE == 0) {
          int btok = m >> 12, ntok = m & (NTOK - 1);
          int h = o >> 6,     d    = o & (DDIM - 1);
          out[(((size_t)btok * HEADS + h) * NTOK + ntok) * DDIM + d] = v;
        } else {
          out[(size_t)m * N + o] = v + bias[o];
        }
      }
    }
  }
}

// fp32 -> bf16 bulk convert (4 elems/thread)
__global__ __launch_bounds__(256)
void f32_to_bf16_kernel(const float* __restrict__ src,
                        unsigned short* __restrict__ dst, int n4) {
  int i = blockIdx.x * 256 + threadIdx.x;
  if (i < n4) {
    float4 v = ((const float4*)src)[i];
    uint2 p;
    p.x = (unsigned)f2bf(v.x) | ((unsigned)f2bf(v.y) << 16);
    p.y = (unsigned)f2bf(v.z) | ((unsigned)f2bf(v.w) << 16);
    ((uint2*)dst)[i] = p;
  }
}

// scale[bh,d] = 1 / max(sqrt(sum_n W^2), 1e-12)^2
__global__ __launch_bounds__(256)
void col_scale_kernel(const float* __restrict__ W, float* __restrict__ scale) {
  int bh = blockIdx.x;
  const float* Wp = W + (size_t)bh * NTOK * DDIM;
  int d = threadIdx.x & 63, g = threadIdx.x >> 6;
  float ss = 0.f;
  for (int n = g; n < NTOK; n += 4) { float v = Wp[(size_t)n * DDIM + d]; ss += v * v; }
  __shared__ float red[4][64];
  red[g][d] = ss;
  __syncthreads();
  if (threadIdx.x < 64) {
    float t = red[0][d] + red[1][d] + red[2][d] + red[3][d];
    float nrm = fmaxf(sqrtf(t), 1e-12f);
    scale[bh * DDIM + d] = 1.0f / (nrm * nrm);
  }
}

// s[bh,n] = temp[h] * sum_d W^2 * scale   (one wave per token)
__global__ __launch_bounds__(256)
void s_kernel(const float* __restrict__ W, const float* __restrict__ scale,
              const float* __restrict__ temp, float* __restrict__ s) {
  int bh   = blockIdx.x;
  int n    = blockIdx.y * 8 + (threadIdx.x >> 5);
  int lane = threadIdx.x & 31;
  const float* Wp = W + ((size_t)bh * NTOK + n) * DDIM;
  const float* sc = scale + bh * DDIM;
  float v0 = Wp[lane], v1 = Wp[lane + 32];
  float acc = v0 * v0 * sc[lane] + v1 * v1 * sc[lane + 32];
  #pragma unroll
  for (int k = 16; k >= 1; k >>= 1) acc += __shfl_xor(acc, k, 32);
  if (lane == 0) s[(size_t)bh * NTOK + n] = acc * temp[bh & (HEADS - 1)];
}

__global__ void zero_kernel(float* __restrict__ p) { p[threadIdx.x] = 0.0f; }

// softmax over heads per token; pi out; sum_n pi per (b,h) via LDS + global atomics
__global__ __launch_bounds__(256)
void softmax_kernel(const float* __restrict__ s, float* __restrict__ pi,
                    float* __restrict__ sumpi) {
  int b = blockIdx.x >> 4, chunk = blockIdx.x & 15;
  int n = chunk * 256 + threadIdx.x;
  float sv[HEADS], mx = -3.4e38f;
  #pragma unroll
  for (int h = 0; h < HEADS; ++h) {
    sv[h] = s[((size_t)(b * HEADS + h)) * NTOK + n];
    mx = fmaxf(mx, sv[h]);
  }
  float sum = 0.f;
  #pragma unroll
  for (int h = 0; h < HEADS; ++h) { sv[h] = __expf(sv[h] - mx); sum += sv[h]; }
  float inv = 1.0f / sum;
  __shared__ float part[HEADS];
  if (threadIdx.x < HEADS) part[threadIdx.x] = 0.0f;
  __syncthreads();
  #pragma unroll
  for (int h = 0; h < HEADS; ++h) {
    float p = sv[h] * inv;
    pi[((size_t)(b * HEADS + h)) * NTOK + n] = p;
    atomicAdd(&part[h], p);                       // ds_add_f32
  }
  __syncthreads();
  if (threadIdx.x < HEADS)
    atomicAdd(&sumpi[b * HEADS + threadIdx.x], part[threadIdx.x]);
}

// attn[bh,d] = 1/(1 + (1/(sumpi+1e-8)) * sum_n pi * W^2)
__global__ __launch_bounds__(256)
void attn_kernel(const float* __restrict__ W, const float* __restrict__ pi,
                 const float* __restrict__ sumpi, float* __restrict__ attn) {
  int bh = blockIdx.x;
  float inv_sp = 1.0f / (sumpi[bh] + 1e-8f);
  const float* Wp = W + (size_t)bh * NTOK * DDIM;
  const float* Pp = pi + (size_t)bh * NTOK;
  int d = threadIdx.x & 63, g = threadIdx.x >> 6;
  float acc = 0.f;
  for (int n = g; n < NTOK; n += 4) {
    float p = Pp[n];
    float v = Wp[(size_t)n * DDIM + d];
    acc += p * v * v;
  }
  __shared__ float red[4][64];
  red[g][d] = acc;
  __syncthreads();
  if (threadIdx.x < 64) {
    float dots = (red[0][d] + red[1][d] + red[2][d] + red[3][d]) * inv_sp;
    attn[bh * DDIM + d] = 1.0f / (1.0f + dots);
  }
}

// out_pre[b,n,h*64+d] = bf16( -W[b,h,n,d] * pi[b,h,n] * attn[b,h,d] )
__global__ __launch_bounds__(256)
void outpre_kernel(const float* __restrict__ W, const float* __restrict__ pi,
                   const float* __restrict__ attn,
                   unsigned short* __restrict__ op) {
  size_t idx = (size_t)blockIdx.x * 256 + threadIdx.x;
  int d  = (int)(idx & (DDIM - 1));
  int n  = (int)((idx >> 6) & (NTOK - 1));
  int bh = (int)(idx >> 18);
  int h = bh & (HEADS - 1), b = bh >> 4;
  float v = -W[idx] * pi[(size_t)bh * NTOK + n] * attn[bh * DDIM + d];
  op[((size_t)(b * NTOK + n)) * CDIM + h * DDIM + d] = f2bf(v);
}

extern "C" void kernel_launch(void* const* d_in, const int* in_sizes, int n_in,
                              void* d_out, int out_size, void* d_ws, size_t ws_size,
                              hipStream_t stream) {
  const float* x     = (const float*)d_in[0];
  const float* qkv_w = (const float*)d_in[1];
  const float* temp  = (const float*)d_in[2];
  const float* out_w = (const float*)d_in[3];
  const float* out_b = (const float*)d_in[4];
  float* y = (float*)d_out;

  char* ws = (char*)d_ws;
  const size_t NELEM  = (size_t)BATCH * NTOK * CDIM;          // 16M
  const size_t WBYTES = NELEM * sizeof(float);                // 64 MB
  const size_t HBYTES = NELEM * sizeof(unsigned short);       // 32 MB
  const size_t QBYTES = (size_t)CDIM * CDIM * 2;              // 2 MB
  const size_t SBYTES = (size_t)BATCH * HEADS * NTOK * 4;     // 1 MB

  float*          W     = (float*)(ws);                                 // 64 MB
  unsigned short* OPbf  = (unsigned short*)(ws + WBYTES);               // 32 MB
  unsigned short* Xbf   = (unsigned short*)(ws + WBYTES + HBYTES);      // 32 MB
  unsigned short* Qbf   = (unsigned short*)(ws + WBYTES + 2 * HBYTES);  // 2 MB
  unsigned short* OWbf  = (unsigned short*)(ws + WBYTES + 2 * HBYTES + QBYTES);
  char* tail = ws + WBYTES + 2 * HBYTES + 2 * QBYTES;
  float* Sbuf  = (float*)(tail);
  float* Pi    = (float*)(tail + SBYTES);
  float* Scale = (float*)(tail + 2 * SBYTES);
  float* Attn  = (float*)(tail + 2 * SBYTES + 16384);
  float* SumPi = (float*)(tail + 2 * SBYTES + 32768);

  const int M = BATCH * NTOK, N = CDIM, K = CDIM;
  dim3 gGemm(N / BN, M / BM);

  // pre-convert GEMM inputs to bf16 (memory-bound, one pass each)
  f32_to_bf16_kernel<<<(int)(NELEM / 4 / 256), 256, 0, stream>>>(x, Xbf, (int)(NELEM / 4));
  f32_to_bf16_kernel<<<(CDIM * CDIM / 4) / 256, 256, 0, stream>>>(qkv_w, Qbf, CDIM * CDIM / 4);
  f32_to_bf16_kernel<<<(CDIM * CDIM / 4) / 256, 256, 0, stream>>>(out_w, OWbf, CDIM * CDIM / 4);

  gemm_bf16_wmma<0><<<gGemm, 256, 0, stream>>>(Xbf, Qbf, W, nullptr, M, N, K);
  col_scale_kernel<<<BATCH * HEADS, 256, 0, stream>>>(W, Scale);
  s_kernel<<<dim3(BATCH * HEADS, NTOK / 8), 256, 0, stream>>>(W, Scale, temp, Sbuf);
  zero_kernel<<<1, BATCH * HEADS, 0, stream>>>(SumPi);
  softmax_kernel<<<BATCH * (NTOK / 256), 256, 0, stream>>>(Sbuf, Pi, SumPi);
  attn_kernel<<<BATCH * HEADS, 256, 0, stream>>>(W, Pi, SumPi, Attn);
  outpre_kernel<<<(int)(NELEM / 256), 256, 0, stream>>>(W, Pi, Attn, OPbf);
  gemm_bf16_wmma<1><<<gGemm, 256, 0, stream>>>(OPbf, OWbf, y, out_b, M, N, K);
}
